// PyramidMixedCausalAttention_34076270526848
// MI455X (gfx1250) — matrix operand order repaired and checked by reading, not compiled, not run on hardware
//
#include <hip/hip_runtime.h>
#include <stdint.h>

#define DD    1024
#define HH    16
#define DHD   64
#define LNSC  64
#define BB    2
#define LL    2048
#define SLEN  (LL - LNSC)      // 1984
#define MROWS (BB * LL)        // 4096

typedef __attribute__((ext_vector_type(16))) __bf16 v16bf;
typedef __attribute__((ext_vector_type(8)))  float  v8f;

union FragBF { uint4 u2[2]; v16bf v; };

__device__ __forceinline__ unsigned short f2bf(float f) {
    union { float f; uint32_t u; } c; c.f = f;
    uint32_t u = c.u;
    if ((u & 0x7fffffffu) > 0x7f800000u) return (unsigned short)((u >> 16) | 0x0040u); // NaN
    uint32_t r = u + 0x7fffu + ((u >> 16) & 1u);                                       // RNE
    return (unsigned short)(r >> 16);
}

__device__ __forceinline__ v8f vzero8() {
    v8f z;
#pragma unroll
    for (int i = 0; i < 8; ++i) z[i] = 0.0f;
    return z;
}

// ---------------------------------------------------------------------------
// DPP16 16-lane reductions (stay inside each 16-lane row; halves of the
// wave32 remain independent).  VALU-only: no LDS, no DScnt waits.
//   0xB1 = quad_perm(1,0,3,2)  -> xor 1
//   0x4E = quad_perm(2,3,0,1)  -> xor 2
//   0x141 = row_half_mirror    -> combine quads within 8
//   0x140 = row_mirror         -> combine 8-groups within 16
// ---------------------------------------------------------------------------
template<int CTRL>
__device__ __forceinline__ float dpp_movf(float x) {
    int xi = __builtin_bit_cast(int, x);
    int r  = __builtin_amdgcn_update_dpp(0, xi, CTRL, 0xF, 0xF, true);
    return __builtin_bit_cast(float, r);
}
__device__ __forceinline__ float redmax16(float v) {
    v = fmaxf(v, dpp_movf<0xB1>(v));
    v = fmaxf(v, dpp_movf<0x4E>(v));
    v = fmaxf(v, dpp_movf<0x141>(v));
    v = fmaxf(v, dpp_movf<0x140>(v));
    return v;
}
__device__ __forceinline__ float redsum16(float v) {
    v += dpp_movf<0xB1>(v);
    v += dpp_movf<0x4E>(v);
    v += dpp_movf<0x141>(v);
    v += dpp_movf<0x140>(v);
    return v;
}

// ---------------------------------------------------------------------------
// fp32 -> bf16 conversion, 4-wide
// ---------------------------------------------------------------------------
__global__ void cvt4_f32_bf16(const float4* __restrict__ s, ushort4* __restrict__ d, int n4) {
    int i = blockIdx.x * blockDim.x + threadIdx.x;
    if (i < n4) {
        float4 v = s[i];
        ushort4 o;
        o.x = f2bf(v.x); o.y = f2bf(v.y); o.z = f2bf(v.z); o.w = f2bf(v.w);
        d[i] = o;
    }
}

// ---------------------------------------------------------------------------
// bf16 GEMM:  Y[M=4096][N=1024] = A[M][K=1024] @ W[N][K]^T  (fp32 accum)
// MODE 0: store bf16 row-major   MODE 1: store bf16 into Vt[B][H][DH][L]
// MODE 2: store fp32 row-major (final output)
// Block = 256 thr (8 waves), block tile 128x128, wave tile 32x64 (2x4 WMMA).
// ---------------------------------------------------------------------------
template<int MODE>
__global__ __launch_bounds__(256)
void gemm_bf16(const unsigned short* __restrict__ A,
               const unsigned short* __restrict__ W,
               void* __restrict__ out)
{
    const int K = DD, N = DD;
    const int lane  = threadIdx.x & 31;
    const int wave  = threadIdx.x >> 5;
    const int waveM = wave >> 1;          // 0..3
    const int waveN = wave & 1;           // 0..1
    const int l15   = lane & 15;
    const int lh    = lane >> 4;          // 0/1
    const int rowBase = blockIdx.y * 128 + waveM * 32;
    const int colBase = blockIdx.x * 128 + waveN * 64;

    v8f acc[2][4];
#pragma unroll
    for (int i = 0; i < 2; ++i)
#pragma unroll
        for (int j = 0; j < 4; ++j) acc[i][j] = vzero8();

    const unsigned short* aptr = A + (size_t)(rowBase + l15) * K + lh * 16;  // A chunk base
    const unsigned short* bptr = W + (size_t)(colBase + l15) * K + lh * 16;  // B chunk base

    for (int k = 0; k < K; k += 32) {
        FragBF a[2], b[4];
#pragma unroll
        for (int i = 0; i < 2; ++i) {
            const unsigned short* p = aptr + (size_t)i * 16 * K + k;
            a[i].u2[0] = *(const uint4*)(p);        // K = k + lh*16 .. +8
            a[i].u2[1] = *(const uint4*)(p + 8);    // K = k + lh*16 + 8 .. +8
        }
#pragma unroll
        for (int j = 0; j < 4; ++j) {
            const unsigned short* p = bptr + (size_t)j * 16 * K + k;
            const uint4* q = (const uint4*)p;       // K = k + lh*16 .. +16 (contig)
            b[j].u2[0] = q[0];
            b[j].u2[1] = q[1];
        }
#pragma unroll
        for (int i = 0; i < 2; ++i)
#pragma unroll
            for (int j = 0; j < 4; ++j)
                acc[i][j] = __builtin_amdgcn_wmma_f32_16x16x32_bf16(
                    false, a[i].v, false, b[j].v, (short)0, acc[i][j], false, false);
    }

#pragma unroll
    for (int i = 0; i < 2; ++i)
#pragma unroll
        for (int j = 0; j < 4; ++j)
#pragma unroll
            for (int e = 0; e < 8; ++e) {
                int r = rowBase + i * 16 + e + lh * 8;
                int c = colBase + j * 16 + l15;
                float v = acc[i][j][e];
                if (MODE == 2) {
                    ((float*)out)[(size_t)r * N + c] = v;
                } else if (MODE == 0) {
                    ((unsigned short*)out)[(size_t)r * N + c] = f2bf(v);
                } else {  // Vt[b][h][dh][L]
                    int bI = r >> 11, l = r & (LL - 1);
                    int h = c >> 6, dh = c & 63;
                    ((unsigned short*)out)[(((size_t)bI * HH + h) * DHD + dh) * LL + l] = f2bf(v);
                }
            }
}

// ---------------------------------------------------------------------------
// Non-shared projections: qT[b,t,e] = sum_d x[b,SLEN+t,d] * W[t,d,e]
// Pure streaming kernel (768 MB of weights read once) — the roofline floor.
// grid (64 t, 3 which), 256 thr; each thread owns 4 consecutive e.
// global_prefetch_b8 issued 256 KB (64 rows) ahead once per 8-row chunk.
// ---------------------------------------------------------------------------
__global__ __launch_bounds__(256)
void ns_proj(const float* __restrict__ x,
             const float* __restrict__ Wq, const float* __restrict__ Wk,
             const float* __restrict__ Wv,
             unsigned short* __restrict__ Qb, unsigned short* __restrict__ Kb,
             unsigned short* __restrict__ Vt)
{
    __shared__ float xs[2][DD];
    const int t = blockIdx.x, which = blockIdx.y, tid = threadIdx.x;
    for (int i = tid; i < DD; i += 256) {
        xs[0][i] = x[((size_t)(0 * LL + SLEN + t)) * DD + i];
        xs[1][i] = x[((size_t)(1 * LL + SLEN + t)) * DD + i];
    }
    __syncthreads();

    const float* W = (which == 0) ? Wq : (which == 1) ? Wk : Wv;
    const float4* W4 = (const float4*)(W + (size_t)t * DD * DD);
    float a0x=0,a0y=0,a0z=0,a0w=0, a1x=0,a1y=0,a1z=0,a1w=0;
    for (int d8 = 0; d8 < DD; d8 += 8) {
        // keep the HBM stream ~256KB (64 rows) ahead of the consume point
        if (d8 + 64 < DD)
            __builtin_prefetch((const void*)&W4[(size_t)(d8 + 64) * (DD / 4) + tid], 0, 0);
#pragma unroll
        for (int u = 0; u < 8; ++u) {
            const int d = d8 + u;
            float4 w = W4[(size_t)d * (DD / 4) + tid];
            float x0 = xs[0][d], x1 = xs[1][d];
            a0x += x0 * w.x; a0y += x0 * w.y; a0z += x0 * w.z; a0w += x0 * w.w;
            a1x += x1 * w.x; a1y += x1 * w.y; a1z += x1 * w.z; a1w += x1 * w.w;
        }
    }
    const int e0 = tid * 4;
    float r[2][4] = {{a0x,a0y,a0z,a0w},{a1x,a1y,a1z,a1w}};
#pragma unroll
    for (int b = 0; b < 2; ++b)
#pragma unroll
        for (int c = 0; c < 4; ++c) {
            int e = e0 + c;
            unsigned short v = f2bf(r[b][c]);
            if (which == 0)      Qb[((size_t)(b * LL + SLEN + t)) * DD + e] = v;
            else if (which == 1) Kb[((size_t)(b * LL + SLEN + t)) * DD + e] = v;
            else Vt[(((size_t)b * HH + (e >> 6)) * DHD + (e & 63)) * LL + (SLEN + t)] = v;
        }
}

// ---------------------------------------------------------------------------
// Flash attention, causal, bf16 WMMA with fp32 accumulation.
// grid (L/64, B*H), block 128 (4 waves); wave owns 16 query rows.
// Key blocks of 32 -> P tile (16x32) is exactly one bf16 A-fragment.
// Softmax row statistics reduced with DPP16 (no LDS traffic).
// ---------------------------------------------------------------------------
__global__ __launch_bounds__(128)
void attn_fwd(const unsigned short* __restrict__ Qb,
              const unsigned short* __restrict__ Kb,
              const unsigned short* __restrict__ Vt,
              unsigned short* __restrict__ Ob)
{
    __shared__ __align__(16) unsigned short pSh[4][16][32];
    const int lane = threadIdx.x & 31, wave = threadIdx.x >> 5;
    const int l15 = lane & 15, lh = lane >> 4;
    const int bh = blockIdx.y, bI = bh >> 4, h = bh & 15;
    const int qBase = blockIdx.x * 64 + wave * 16;
    const float scale = 0.125f;  // DH^-0.5

    // Q fragments (16 rows x 64 K, split into two 16x32 A-frags)
    FragBF aQ[2];
    {
        const unsigned short* qp =
            Qb + ((size_t)(bI * LL) + qBase + l15) * DD + h * DHD + lh * 16;
        aQ[0].u2[0] = *(const uint4*)(qp);
        aQ[0].u2[1] = *(const uint4*)(qp + 8);
        aQ[1].u2[0] = *(const uint4*)(qp + 32);
        aQ[1].u2[1] = *(const uint4*)(qp + 40);
    }

    v8f acc[4];
#pragma unroll
    for (int tdh = 0; tdh < 4; ++tdh) acc[tdh] = vzero8();
    float mrow[8], lrow[8];
#pragma unroll
    for (int e = 0; e < 8; ++e) { mrow[e] = -3.0e38f; lrow[e] = 0.0f; }

    const unsigned short* kbase = Kb + (size_t)(bI * LL) * DD + h * DHD + lh * 16;
    const unsigned short* vbase = Vt + (((size_t)bI * HH + h) * DHD + l15) * LL + lh * 16;

    const int nkb = (qBase + 16 + 31) >> 5;  // causal key-block bound
    for (int j = 0; j < nkb; ++j) {
        const int key0 = j << 5;
        // ---- S = Q @ K^T (two 16-key subtiles) ----
        v8f s[2]; s[0] = vzero8(); s[1] = vzero8();
#pragma unroll
        for (int nt = 0; nt < 2; ++nt) {
            const unsigned short* kp = kbase + (size_t)(key0 + nt * 16 + l15) * DD;
            FragBF b0, b1;
            b0.u2[0] = *(const uint4*)(kp);      b0.u2[1] = *(const uint4*)(kp + 8);
            b1.u2[0] = *(const uint4*)(kp + 32); b1.u2[1] = *(const uint4*)(kp + 40);
            s[nt] = __builtin_amdgcn_wmma_f32_16x16x32_bf16(
                false, aQ[0].v, false, b0.v, (short)0, s[nt], false, false);
            s[nt] = __builtin_amdgcn_wmma_f32_16x16x32_bf16(
                false, aQ[1].v, false, b1.v, (short)0, s[nt], false, false);
        }
        // ---- scale + causal mask + online softmax (DPP reductions) ----
#pragma unroll
        for (int e = 0; e < 8; ++e) {
            const int q = qBase + e + lh * 8;
            float s0 = s[0][e] * scale; if (key0 + l15 > q)      s0 = -1.0e9f;
            float s1 = s[1][e] * scale; if (key0 + 16 + l15 > q) s1 = -1.0e9f;
            float vmx = redmax16(fmaxf(s0, s1));
            float nm  = fmaxf(mrow[e], vmx);
            float al  = __expf(mrow[e] - nm);
            mrow[e] = nm;
            float p0 = __expf(s0 - nm);
            float p1 = __expf(s1 - nm);
            float rs = redsum16(p0 + p1);
            lrow[e] = lrow[e] * al + rs;
#pragma unroll
            for (int tdh = 0; tdh < 4; ++tdh) acc[tdh][e] *= al;
            const int prow = e + lh * 8;
            pSh[wave][prow][l15]      = f2bf(p0);
            pSh[wave][prow][16 + l15] = f2bf(p1);
        }
        asm volatile("s_wait_dscnt 0" ::: "memory");  // wave-local LDS store->load
        // ---- re-read P as a 16x32 bf16 A-fragment ----
        FragBF aP;
        {
            const unsigned short* pp = &pSh[wave][l15][lh * 16];
            aP.u2[0] = *(const uint4*)(pp);
            aP.u2[1] = *(const uint4*)(pp + 8);
        }
        // ---- O += P @ V  (V pre-transposed: contiguous B-frag loads) ----
#pragma unroll
        for (int tdh = 0; tdh < 4; ++tdh) {
            const unsigned short* vp = vbase + (size_t)(tdh * 16) * LL + key0;
            FragBF bV;
            bV.u2[0] = *(const uint4*)(vp);
            bV.u2[1] = *(const uint4*)(vp + 8);
            acc[tdh] = __builtin_amdgcn_wmma_f32_16x16x32_bf16(
                false, aP.v, false, bV.v, (short)0, acc[tdh], false, false);
        }
    }

    // ---- normalize + store O (bf16, row-major [B][L][D]) ----
#pragma unroll
    for (int tdh = 0; tdh < 4; ++tdh)
#pragma unroll
        for (int e = 0; e < 8; ++e) {
            const int q = qBase + e + lh * 8;
            const int c = h * DHD + tdh * 16 + l15;
            Ob[((size_t)(bI * LL) + q) * DD + c] = f2bf(acc[tdh][e] / lrow[e]);
        }
}

// ---------------------------------------------------------------------------
extern "C" void kernel_launch(void* const* d_in, const int* in_sizes, int n_in,
                              void* d_out, int out_size, void* d_ws, size_t ws_size,
                              hipStream_t stream) {
    (void)in_sizes; (void)n_in; (void)out_size; (void)ws_size;
    const float* x    = (const float*)d_in[0];
    const float* WqS  = (const float*)d_in[1];
    const float* WkS  = (const float*)d_in[2];
    const float* WvS  = (const float*)d_in[3];
    const float* WqNS = (const float*)d_in[4];
    const float* WkNS = (const float*)d_in[5];
    const float* WvNS = (const float*)d_in[6];
    const float* Wo   = (const float*)d_in[7];

    char* ws = (char*)d_ws;
    const size_t MB = 1024 * 1024;
    unsigned short* XB  = (unsigned short*)(ws + 0);        // x bf16          8 MB
    unsigned short* WQB = (unsigned short*)(ws + 8  * MB);  // WqS bf16        2 MB
    unsigned short* WKB = (unsigned short*)(ws + 10 * MB);
    unsigned short* WVB = (unsigned short*)(ws + 12 * MB);
    unsigned short* WOB = (unsigned short*)(ws + 14 * MB);
    unsigned short* QB  = (unsigned short*)(ws + 16 * MB);  // Q bf16          8 MB
    unsigned short* KB  = (unsigned short*)(ws + 24 * MB);  // K bf16          8 MB
    unsigned short* VT  = (unsigned short*)(ws + 32 * MB);  // V^T bf16        8 MB
    unsigned short* OB  = (unsigned short*)(ws + 40 * MB);  // attn out bf16   8 MB

    // 1) fp32 -> bf16
    {
        int n4 = (BB * LL * DD) / 4;
        cvt4_f32_bf16<<<(n4 + 255) / 256, 256, 0, stream>>>((const float4*)x, (ushort4*)XB, n4);
        int w4 = (DD * DD) / 4;
        cvt4_f32_bf16<<<(w4 + 255) / 256, 256, 0, stream>>>((const float4*)WqS, (ushort4*)WQB, w4);
        cvt4_f32_bf16<<<(w4 + 255) / 256, 256, 0, stream>>>((const float4*)WkS, (ushort4*)WKB, w4);
        cvt4_f32_bf16<<<(w4 + 255) / 256, 256, 0, stream>>>((const float4*)WvS, (ushort4*)WVB, w4);
        cvt4_f32_bf16<<<(w4 + 255) / 256, 256, 0, stream>>>((const float4*)Wo,  (ushort4*)WOB, w4);
    }

    // 2) shared projections (full 4096 rows; NS kernel overwrites last 64 per batch)
    dim3 gg(DD / 128, MROWS / 128);  // (8, 32)
    gemm_bf16<0><<<gg, 256, 0, stream>>>(XB, WQB, QB);
    gemm_bf16<0><<<gg, 256, 0, stream>>>(XB, WKB, KB);
    gemm_bf16<1><<<gg, 256, 0, stream>>>(XB, WVB, VT);   // transposed store

    // 3) non-shared per-position projections (BW-bound: 768 MB streamed once)
    ns_proj<<<dim3(LNSC, 3), 256, 0, stream>>>(x, WqNS, WkNS, WvNS, QB, KB, VT);

    // 4) causal flash attention
    attn_fwd<<<dim3(LL / 64, BB * HH), 128, 0, stream>>>(QB, KB, VT, OB);

    // 5) output projection -> fp32
    gemm_bf16<2><<<gg, 256, 0, stream>>>(OB, WOB, d_out);
}
